// SwinTransformerBlock_21758304322022
// MI455X (gfx1250) — compile-verified, matrix operand order checked by reading, not run on hardware
//
#include <hip/hip_runtime.h>

typedef __attribute__((ext_vector_type(16))) __bf16 v16bf;
typedef __attribute__((ext_vector_type(8)))  float  v8f;
typedef __attribute__((ext_vector_type(4)))  int    v4i;

#define M_TOK   200704      // 64 * 56 * 56
#define C_DIM   128
#define MLP_DIM 512
#define QKV_DIM 384

#if defined(__gfx1250__) && __has_builtin(__builtin_amdgcn_global_load_async_to_lds_b128)
#define USE_ASYNC_COPY 1
#else
#define USE_ASYNC_COPY 0
#endif

#if USE_ASYNC_COPY
#if __has_builtin(__builtin_amdgcn_s_wait_asynccnt)
#define WAIT_ASYNC(n) __builtin_amdgcn_s_wait_asynccnt(n)
#else
#define WAIT_ASYNC(n) asm volatile("s_wait_asynccnt %0" ::"i"(n) : "memory")
#endif
#else
#define WAIT_ASYNC(n)
#endif

#define LDS_STRIDE 40   // 80B rows: 16B-aligned for b128 LDS writes, bank-conflict-free frags

// ---------------------------------------------------------------- helpers

__device__ __forceinline__ unsigned short f2bf(float f) {
    union { float f; unsigned u; } cv; cv.f = f;
    unsigned r = cv.u + 0x7FFFu + ((cv.u >> 16) & 1u);   // RNE
    return (unsigned short)(r >> 16);
}

__device__ __forceinline__ float gelu_tanh(float x) {
    float x3 = x * x * x;
    return 0.5f * x * (1.0f + tanhf(0.7978845608028654f * (x + 0.044715f * x3)));
}

// 16-bit A/B fragment K pattern (ISA 7.12.2): vgpr v holds K pair
// k = (v/4)*16 + half*8 + (v%4)*2  (pairs contiguous -> one 4B LDS read)
__device__ __forceinline__ int kpat(int v, int hl) {
    return ((v >> 2) << 4) + (hl << 3) + ((v & 3) << 1);
}

__device__ __forceinline__ v16bf load_frag(const unsigned short* rowp, int koff, int hl) {
    union { v16bf v; unsigned u[8]; } f;
#pragma unroll
    for (int v = 0; v < 8; ++v)
        f.u[v] = *(const unsigned*)(rowp + koff + kpat(v, hl));
    return f.v;
}

// ---------------------------------------------------------------- LayerNorm -> bf16

__global__ __launch_bounds__(256) void ln_kernel(
    const float* __restrict__ x, const float* __restrict__ g,
    const float* __restrict__ b, unsigned short* __restrict__ out, int M)
{
    int row  = blockIdx.x * 8 + (threadIdx.x >> 5);
    int lane = threadIdx.x & 31;
    if (row >= M) return;
    float4 v = ((const float4*)(x + (size_t)row * C_DIM))[lane];
    float s1 = v.x + v.y + v.z + v.w;
    float s2 = v.x * v.x + v.y * v.y + v.z * v.z + v.w * v.w;
#pragma unroll
    for (int m = 1; m < 32; m <<= 1) {
        s1 += __shfl_xor(s1, m);
        s2 += __shfl_xor(s2, m);
    }
    float mu  = s1 * (1.0f / 128.0f);
    float var = s2 * (1.0f / 128.0f) - mu * mu;
    float rs  = rsqrtf(var + 1e-5f);
    float4 gv = ((const float4*)g)[lane];
    float4 bv = ((const float4*)b)[lane];
    unsigned short o0 = f2bf((v.x - mu) * rs * gv.x + bv.x);
    unsigned short o1 = f2bf((v.y - mu) * rs * gv.y + bv.y);
    unsigned short o2 = f2bf((v.z - mu) * rs * gv.z + bv.z);
    unsigned short o3 = f2bf((v.w - mu) * rs * gv.w + bv.w);
    uint2 pk;
    pk.x = (unsigned)o0 | ((unsigned)o1 << 16);
    pk.y = (unsigned)o2 | ((unsigned)o3 << 16);
    ((uint2*)(out + (size_t)row * C_DIM))[lane] = pk;
}

// ---------------------------------------------------------------- fp32 -> bf16 (transposing) convert
// in: [K,N] fp32 row-major  ->  out: [N,K] bf16 row-major (W^T)

__global__ void cvt_t_kernel(const float* __restrict__ in, unsigned short* __restrict__ out,
                             int K, int N)
{
    int i = blockIdx.x * 256 + threadIdx.x;
    if (i < K * N) {
        int k = i / N, n = i % N;
        out[(size_t)n * K + k] = f2bf(in[i]);
    }
}

// ---------------------------------------------------------------- WMMA GEMM
// C[M,N] = act(A[M,K]bf16 @ W^T[N,K]bf16 + bias) (+ res fp32) -> bf16 or fp32
// Block tile 128x128, 8 waves (4x2), wave tile 32x64 (2x4 wmma 16x16x32 bf16).
// Double-buffered LDS staging via GLOBAL_LOAD_ASYNC_TO_LDS_B128 when available.

__device__ __forceinline__ void stage_tile(const unsigned short* __restrict__ g,
                                           unsigned short* l, int ld, int k0,
                                           int base, int tid)
{
#pragma unroll
    for (int c = 0; c < 2; ++c) {
        int ch = tid + c * 256;
        int r = ch >> 2, col = (ch & 3) << 3;           // 128 rows x 32 cols, 8-elem chunks
        const unsigned short* src = g + (size_t)(base + r) * ld + k0 + col;
        unsigned short* dst = l + r * LDS_STRIDE + col;
#if USE_ASYNC_COPY
        __builtin_amdgcn_global_load_async_to_lds_b128(
            (__attribute__((address_space(1))) v4i*)src,
            (__attribute__((address_space(3))) v4i*)dst, 0, 0);
#else
        uint4 d = *(const uint4*)src;
        unsigned* dw = (unsigned*)dst;
        dw[0] = d.x; dw[1] = d.y; dw[2] = d.z; dw[3] = d.w;
#endif
    }
}

template<int ACT, int RES, int OBF, int K>
__global__ __launch_bounds__(256) void gemm_bf16(
    const unsigned short* __restrict__ A, const unsigned short* __restrict__ B,
    const float* __restrict__ bias, const float* __restrict__ res,
    float* __restrict__ outF, unsigned short* __restrict__ outB,
    int M, int N)
{
    __shared__ unsigned short As[2][128 * LDS_STRIDE];
    __shared__ unsigned short Bs[2][128 * LDS_STRIDE];

    const int bm = blockIdx.y * 128, bn = blockIdx.x * 128;
    const int tid = threadIdx.x, lane = tid & 31, wid = tid >> 5;
    const int wm = (wid >> 1) * 32, wn = (wid & 1) * 64;
    const int hl = lane >> 4, l16 = lane & 15;
    const int KT = K / 32;

    v8f acc[2][4] = {};

    stage_tile(A, As[0], K, 0, bm, tid);
    stage_tile(B, Bs[0], K, 0, bn, tid);

#pragma unroll
    for (int kt = 0; kt < KT; ++kt) {
        if (kt + 1 < KT) {
            stage_tile(A, As[(kt + 1) & 1], K, (kt + 1) * 32, bm, tid);
            stage_tile(B, Bs[(kt + 1) & 1], K, (kt + 1) * 32, bn, tid);
        }
#if USE_ASYNC_COPY
        if (kt + 1 < KT) { WAIT_ASYNC(4); } else { WAIT_ASYNC(0); }
#endif
        __syncthreads();

        const unsigned short* curA = As[kt & 1];
        const unsigned short* curB = Bs[kt & 1];
        v16bf af[2], bfrag[4];
#pragma unroll
        for (int mt = 0; mt < 2; ++mt)
            af[mt] = load_frag(curA + (wm + mt * 16 + l16) * LDS_STRIDE, 0, hl);
#pragma unroll
        for (int nt = 0; nt < 4; ++nt)
            bfrag[nt] = load_frag(curB + (wn + nt * 16 + l16) * LDS_STRIDE, 0, hl);
#pragma unroll
        for (int mt = 0; mt < 2; ++mt)
#pragma unroll
            for (int nt = 0; nt < 4; ++nt)
                acc[mt][nt] = __builtin_amdgcn_wmma_f32_16x16x32_bf16(
                    false, af[mt], false, bfrag[nt], (short)0, acc[mt][nt], false, false);
        __syncthreads();
    }

    // epilogue: bias (+gelu) (+residual), store bf16 or fp32
#pragma unroll
    for (int mt = 0; mt < 2; ++mt)
#pragma unroll
        for (int nt = 0; nt < 4; ++nt) {
            int col = bn + wn + nt * 16 + l16;
            float bsv = bias[col];
#pragma unroll
            for (int v = 0; v < 8; ++v) {
                int row = bm + wm + mt * 16 + hl * 8 + v;
                float val = acc[mt][nt][v] + bsv;
                if (ACT) val = gelu_tanh(val);
                if (RES) val += res[(size_t)row * N + col];
                if (OBF) outB[(size_t)row * N + col] = f2bf(val);
                else     outF[(size_t)row * N + col] = val;
            }
        }
}

// ---------------------------------------------------------------- windowed attention
// One block per (batch, window, head). 4 waves; wave w owns S/O rows 16w..16w+15.
// Roll(shift) + window partition folded into the gather index; scatter un-rolls.

__global__ __launch_bounds__(128) void attn_kernel(
    const unsigned short* __restrict__ qkv, unsigned short* __restrict__ outA, int shift)
{
    __shared__ unsigned short Qs[64 * 34];
    __shared__ unsigned short Ks[64 * 34];
    __shared__ unsigned short Vt[32 * 66];   // transposed: [d][token]
    __shared__ unsigned short Ps[64 * 66];
    __shared__ int ti[64];

    const int blk = blockIdx.x;
    const int h = blk & 3, w = (blk >> 2) & 63, b = blk >> 8;
    const int tid = threadIdx.x, lane = tid & 31, wid = tid >> 5;

    if (tid < 64) {
        int n = tid, ni = n / 7, nj = n % 7;
        int wi = w >> 3, wj = w & 7;
        int ri = (wi * 7 + ni + 56 - shift) % 56;
        int rj = (wj * 7 + nj + 56 - shift) % 56;
        ti[tid] = (n < 49) ? (b * 3136 + ri * 56 + rj) : 0;
    }
    __syncthreads();

    // stage Q,K (row major, padded) and V (transposed) for this head
    {
        int r = tid >> 1, cb = (tid & 1) * 16;
        if (r < 49) {
            size_t base = (size_t)ti[r] * QKV_DIM + h * 32 + cb;
            uint4 q0 = *(const uint4*)(qkv + base);
            uint4 q1 = *(const uint4*)(qkv + base + 8);
            uint4 k0 = *(const uint4*)(qkv + base + 128);
            uint4 k1 = *(const uint4*)(qkv + base + 136);
            uint4 v0 = *(const uint4*)(qkv + base + 256);
            uint4 v1 = *(const uint4*)(qkv + base + 264);
            unsigned* qd = (unsigned*)(Qs + r * 34 + cb);
            qd[0] = q0.x; qd[1] = q0.y; qd[2] = q0.z; qd[3] = q0.w;
            qd[4] = q1.x; qd[5] = q1.y; qd[6] = q1.z; qd[7] = q1.w;
            unsigned* kd = (unsigned*)(Ks + r * 34 + cb);
            kd[0] = k0.x; kd[1] = k0.y; kd[2] = k0.z; kd[3] = k0.w;
            kd[4] = k1.x; kd[5] = k1.y; kd[6] = k1.z; kd[7] = k1.w;
            unsigned short vv[16];
            *(uint4*)(vv)     = v0;
            *(uint4*)(vv + 8) = v1;
#pragma unroll
            for (int j = 0; j < 16; ++j) Vt[(cb + j) * 66 + r] = vv[j];
        } else {
#pragma unroll
            for (int j = 0; j < 16; ++j) {
                Qs[r * 34 + cb + j] = 0;
                Ks[r * 34 + cb + j] = 0;
                Vt[(cb + j) * 66 + r] = 0;
            }
        }
    }
    __syncthreads();

    const int hl = lane >> 4, l16 = lane & 15;
    const int rbase = wid * 16;

    // S = Q @ K^T  (K=32, one wmma per 16x16 tile; 4 N tiles)
    v16bf aq = load_frag(Qs + (rbase + l16) * 34, 0, hl);
    v8f s[4] = {};
#pragma unroll
    for (int nt = 0; nt < 4; ++nt) {
        v16bf bk = load_frag(Ks + (nt * 16 + l16) * 34, 0, hl);
        s[nt] = __builtin_amdgcn_wmma_f32_16x16x32_bf16(
            false, aq, false, bk, (short)0, s[nt], false, false);
    }

    // softmax over the 64 (49 valid) columns of each row
    const float scale = 0.17677669529663687f;   // 1/sqrt(32)
#pragma unroll
    for (int v = 0; v < 8; ++v) {
        float mx = -3.0e38f;
#pragma unroll
        for (int nt = 0; nt < 4; ++nt) {
            float val = s[nt][v] * scale;
            if (nt * 16 + l16 >= 49) val = -3.0e38f;
            s[nt][v] = val;
            mx = fmaxf(mx, val);
        }
#pragma unroll
        for (int m = 1; m < 16; m <<= 1) mx = fmaxf(mx, __shfl_xor(mx, m));
        float sum = 0.0f;
#pragma unroll
        for (int nt = 0; nt < 4; ++nt) {
            float e = __expf(s[nt][v] - mx);
            s[nt][v] = e;
            sum += e;
        }
#pragma unroll
        for (int m = 1; m < 16; m <<= 1) sum += __shfl_xor(sum, m);
        float inv = 1.0f / sum;
        int rr = rbase + hl * 8 + v;
#pragma unroll
        for (int nt = 0; nt < 4; ++nt)
            Ps[rr * 66 + nt * 16 + l16] = f2bf(s[nt][v] * inv);
    }
    __syncthreads();

    // O = P @ V  (K=64 -> 2 chunks of 32; 2 N tiles of 16 over HD=32)
    v8f o[2] = {};
#pragma unroll
    for (int c = 0; c < 2; ++c) {
        v16bf ap = load_frag(Ps + (rbase + l16) * 66, c * 32, hl);
#pragma unroll
        for (int nt = 0; nt < 2; ++nt) {
            v16bf bv = load_frag(Vt + (nt * 16 + l16) * 66, c * 32, hl);
            o[nt] = __builtin_amdgcn_wmma_f32_16x16x32_bf16(
                false, ap, false, bv, (short)0, o[nt], false, false);
        }
    }

    // scatter back to token order (un-roll happens automatically: same ti)
#pragma unroll
    for (int nt = 0; nt < 2; ++nt) {
        int d = nt * 16 + l16;
#pragma unroll
        for (int v = 0; v < 8; ++v) {
            int rr = rbase + hl * 8 + v;
            if (rr < 49)
                outA[(size_t)ti[rr] * C_DIM + h * 32 + d] = f2bf(o[nt][v]);
        }
    }
}

// ---------------------------------------------------------------- launch

extern "C" void kernel_launch(void* const* d_in, const int* in_sizes, int n_in,
                              void* d_out, int out_size, void* d_ws, size_t ws_size,
                              hipStream_t stream)
{
    const float* x      = (const float*)d_in[0];
    const float* ln1_g  = (const float*)d_in[1];
    const float* ln1_b  = (const float*)d_in[2];
    const float* w1_qkv = (const float*)d_in[3];
    const float* b1_qkv = (const float*)d_in[4];
    const float* w1_o   = (const float*)d_in[5];
    const float* b1_o   = (const float*)d_in[6];
    const float* ln2_g  = (const float*)d_in[7];
    const float* ln2_b  = (const float*)d_in[8];
    const float* f1_w1  = (const float*)d_in[9];
    const float* f1_b1  = (const float*)d_in[10];
    const float* f1_w2  = (const float*)d_in[11];
    const float* f1_b2  = (const float*)d_in[12];
    const float* ln3_g  = (const float*)d_in[13];
    const float* ln3_b  = (const float*)d_in[14];
    const float* w2_qkv = (const float*)d_in[15];
    const float* b2_qkv = (const float*)d_in[16];
    const float* w2_o   = (const float*)d_in[17];
    const float* b2_o   = (const float*)d_in[18];
    const float* ln4_g  = (const float*)d_in[19];
    const float* ln4_b  = (const float*)d_in[20];
    const float* f2_w1  = (const float*)d_in[21];
    const float* f2_b1  = (const float*)d_in[22];
    const float* f2_w2  = (const float*)d_in[23];
    const float* f2_b2  = (const float*)d_in[24];

    float* out = (float*)d_out;
    const int M = M_TOK;

    // bf16 workspace arena with lifetime reuse
    unsigned short* arena = (unsigned short*)d_ws;
    unsigned short* xn   = arena;                      // M*128 (also attn-out, xn2)
    unsigned short* qkvb = arena + (size_t)M * 128;    // M*384 (also hmid M*512)
    unsigned short* attb = xn;
    unsigned short* hmid = qkvb;
    unsigned short* wts  = arena + (size_t)M * 640;

    unsigned short* wqkv1 = wts;                       // transposed [N,K]
    unsigned short* wo1   = wts + 49152;
    unsigned short* wf11  = wts + 65536;
    unsigned short* wf12  = wts + 131072;
    unsigned short* wqkv2 = wts + 196608;
    unsigned short* wo2   = wts + 245760;
    unsigned short* wf21  = wts + 262144;
    unsigned short* wf22  = wts + 327680;

    // convert+transpose weights fp32 [K,N] -> bf16 [N,K] (deterministic, every call)
    cvt_t_kernel<<<(49152 + 255) / 256, 256, 0, stream>>>(w1_qkv, wqkv1, C_DIM,   QKV_DIM);
    cvt_t_kernel<<<(16384 + 255) / 256, 256, 0, stream>>>(w1_o,   wo1,   C_DIM,   C_DIM);
    cvt_t_kernel<<<(65536 + 255) / 256, 256, 0, stream>>>(f1_w1,  wf11,  C_DIM,   MLP_DIM);
    cvt_t_kernel<<<(65536 + 255) / 256, 256, 0, stream>>>(f1_w2,  wf12,  MLP_DIM, C_DIM);
    cvt_t_kernel<<<(49152 + 255) / 256, 256, 0, stream>>>(w2_qkv, wqkv2, C_DIM,   QKV_DIM);
    cvt_t_kernel<<<(16384 + 255) / 256, 256, 0, stream>>>(w2_o,   wo2,   C_DIM,   C_DIM);
    cvt_t_kernel<<<(65536 + 255) / 256, 256, 0, stream>>>(f2_w1,  wf21,  C_DIM,   MLP_DIM);
    cvt_t_kernel<<<(65536 + 255) / 256, 256, 0, stream>>>(f2_w2,  wf22,  MLP_DIM, C_DIM);

    dim3 blk(256);
    dim3 gQKV(QKV_DIM / 128, M / 128);
    dim3 gC  (C_DIM   / 128, M / 128);
    dim3 gMLP(MLP_DIM / 128, M / 128);

    // ---------- half 1 (no shift) ----------
    ln_kernel<<<M / 8, 256, 0, stream>>>(x, ln1_g, ln1_b, xn, M);
    gemm_bf16<0, 0, 1, C_DIM><<<gQKV, blk, 0, stream>>>(xn, wqkv1, b1_qkv, nullptr, nullptr, qkvb, M, QKV_DIM);
    attn_kernel<<<64 * 64 * 4, 128, 0, stream>>>(qkvb, attb, 0);
    gemm_bf16<0, 1, 0, C_DIM><<<gC, blk, 0, stream>>>(attb, wo1, b1_o, x, out, nullptr, M, C_DIM);
    ln_kernel<<<M / 8, 256, 0, stream>>>(out, ln2_g, ln2_b, xn, M);
    gemm_bf16<1, 0, 1, C_DIM><<<gMLP, blk, 0, stream>>>(xn, wf11, f1_b1, nullptr, nullptr, hmid, M, MLP_DIM);
    gemm_bf16<0, 1, 0, MLP_DIM><<<gC, blk, 0, stream>>>(hmid, wf12, f1_b2, out, out, nullptr, M, C_DIM);

    // ---------- half 2 (shift = 3) ----------
    ln_kernel<<<M / 8, 256, 0, stream>>>(out, ln3_g, ln3_b, xn, M);
    gemm_bf16<0, 0, 1, C_DIM><<<gQKV, blk, 0, stream>>>(xn, wqkv2, b2_qkv, nullptr, nullptr, qkvb, M, QKV_DIM);
    attn_kernel<<<64 * 64 * 4, 128, 0, stream>>>(qkvb, attb, 3);
    gemm_bf16<0, 1, 0, C_DIM><<<gC, blk, 0, stream>>>(attb, wo2, b2_o, out, out, nullptr, M, C_DIM);
    ln_kernel<<<M / 8, 256, 0, stream>>>(out, ln4_g, ln4_b, xn, M);
    gemm_bf16<1, 0, 1, C_DIM><<<gMLP, blk, 0, stream>>>(xn, wf21, f2_b1, nullptr, nullptr, hmid, M, MLP_DIM);
    gemm_bf16<0, 1, 0, MLP_DIM><<<gC, blk, 0, stream>>>(hmid, wf22, f2_b2, out, out, nullptr, M, C_DIM);
}
